// QTypeEmbeddingEmbPred_35845797053098
// MI455X (gfx1250) — matrix operand order repaired and checked by reading, not compiled, not run on hardware
//
#include <hip/hip_runtime.h>

typedef __attribute__((ext_vector_type(16))) __bf16      v16bf;
typedef __attribute__((ext_vector_type(8)))  float       v8f;
typedef __attribute__((ext_vector_type(4)))  unsigned int u32x4;

#define B_DIM   8192
#define IN_DIM  768
#define EMB_DIM 3072
#define V_DIM   8192

static __device__ __forceinline__ unsigned short f2bf(float x) {
    unsigned u = __builtin_bit_cast(unsigned int, x);
    u += 0x7fffu + ((u >> 16) & 1u);      // round-to-nearest-even
    return (unsigned short)(u >> 16);
}

union FragCast { u32x4 u[2]; v16bf v; };

// A-matrix 16x32 bf16 fragment (ISA 7.12.2): lane<16 -> M=lane, K = k..k+7 and k+16..k+23
//                                            lane>=16 -> M=lane-16, K offset +8
static __device__ __forceinline__ v16bf load_frag_a(const unsigned short* base, int ld, int k, int lane) {
    const int half = lane >> 4;
    const int m    = lane & 15;
    const unsigned short* p = base + (size_t)m * ld + k + half * 8;
    FragCast f;
    f.u[0] = *reinterpret_cast<const u32x4*>(p);       // K = k..k+7   (+8 for upper half)
    f.u[1] = *reinterpret_cast<const u32x4*>(p + 16);  // K = k+16..k+23
    return f.v;
}

// B-matrix 32x16 bf16 fragment: lane<16 -> N=lane, K = k..k+7 and k+16..k+23; lane>=16 -> K offset +8
static __device__ __forceinline__ v16bf load_frag_b(const unsigned short* base, int ld, int k, int lane) {
    const int half = lane >> 4;
    const int n    = lane & 15;
    const unsigned short* p = base + (size_t)n * ld + k + half * 8;
    FragCast f;
    f.u[0] = *reinterpret_cast<const u32x4*>(p);       // K = k..k+7
    f.u[1] = *reinterpret_cast<const u32x4*>(p + 16);  // K = k+16..k+23
    return f.v;
}

// ---------------- conversion kernels ----------------
__global__ void cvt_bf16(const float* __restrict__ in, unsigned short* __restrict__ out, int n) {
    int i = blockIdx.x * blockDim.x + threadIdx.x;
    if (i < n) out[i] = f2bf(in[i]);
}

__global__ void cvt_w_transpose(const float* __restrict__ W, unsigned short* __restrict__ Wt) {
    int t = blockIdx.x * blockDim.x + threadIdx.x;   // over IN_DIM*EMB_DIM
    if (t < IN_DIM * EMB_DIM) {
        int i = t / EMB_DIM;       // input-dim row
        int j = t - i * EMB_DIM;   // emb col
        Wt[(size_t)j * IN_DIM + i] = f2bf(W[t]);
    }
}

__global__ void init_amax(unsigned long long* __restrict__ amax) {
    int i = blockIdx.x * blockDim.x + threadIdx.x;
    if (i < B_DIM) amax[i] = 0ull;
}

// ---------------- GEMM1: pred = raw @ W + b  (bf16 WMMA, f32 accum) ----------------
// block = 128m x 128n, 8 waves (2m x 4n), wave tile = 64m x 32n
__global__ __launch_bounds__(256) void gemm1_bias(const unsigned short* __restrict__ rawbf,
                                                  const unsigned short* __restrict__ wtbf,
                                                  const float* __restrict__ bias,
                                                  float* __restrict__ predf,
                                                  unsigned short* __restrict__ predbf) {
    const int lane  = threadIdx.x & 31;
    const int wave  = threadIdx.x >> 5;
    const int mBase = blockIdx.x * 128 + (wave & 1) * 64;
    const int nBase = blockIdx.y * 128 + (wave >> 1) * 32;

    v8f acc[4][2];
#pragma unroll
    for (int mt = 0; mt < 4; ++mt)
#pragma unroll
        for (int nt = 0; nt < 2; ++nt)
            acc[mt][nt] = (v8f){0.f,0.f,0.f,0.f,0.f,0.f,0.f,0.f};

    const unsigned short* aP[4];
    const unsigned short* bP[2];
#pragma unroll
    for (int mt = 0; mt < 4; ++mt) aP[mt] = rawbf + (size_t)(mBase + mt * 16) * IN_DIM;
#pragma unroll
    for (int nt = 0; nt < 2; ++nt) bP[nt] = wtbf + (size_t)(nBase + nt * 16) * IN_DIM;

    for (int k = 0; k < IN_DIM; k += 32) {
        v16bf a[4], b[2];
#pragma unroll
        for (int mt = 0; mt < 4; ++mt) a[mt] = load_frag_a(aP[mt], IN_DIM, k, lane);
#pragma unroll
        for (int nt = 0; nt < 2; ++nt) b[nt] = load_frag_b(bP[nt], IN_DIM, k, lane);
#pragma unroll
        for (int mt = 0; mt < 4; ++mt)
#pragma unroll
            for (int nt = 0; nt < 2; ++nt)
                acc[mt][nt] = __builtin_amdgcn_wmma_f32_16x16x32_bf16(
                    false, a[mt], false, b[nt], (short)0, acc[mt][nt], false, false);
    }

    const int half = lane >> 4, l15 = lane & 15;
#pragma unroll
    for (int nt = 0; nt < 2; ++nt) {
        const int n  = nBase + nt * 16 + l15;
        const float bv = bias[n];
#pragma unroll
        for (int mt = 0; mt < 4; ++mt) {
#pragma unroll
            for (int r = 0; r < 8; ++r) {
                const int m = mBase + mt * 16 + half * 8 + r;
                const float v = acc[mt][nt][r] + bv;
                const size_t o = (size_t)m * EMB_DIM + n;
                predf[o]  = v;
                predbf[o] = f2bf(v);
            }
        }
    }
}

// ---------------- GEMM2: scores = pred @ table^T + per-row argmax ----------------
// block = 128m x 256n, 8 waves (2m x 4n), wave tile = 64m x 64n:
// 16 WMMAs per 8 fragment loads per K-step, 4x reuse of every fragment.
__global__ __launch_bounds__(256) void gemm2_argmax(const unsigned short* __restrict__ predbf,
                                                    const unsigned short* __restrict__ tabbf,
                                                    unsigned long long* __restrict__ amax) {
    const int lane  = threadIdx.x & 31;
    const int wave  = threadIdx.x >> 5;
    const int mBase = blockIdx.x * 128 + (wave & 1) * 64;   // batch rows
    const int nBase = blockIdx.y * 256 + (wave >> 1) * 64;  // codebook rows

    v8f acc[4][4];
#pragma unroll
    for (int mt = 0; mt < 4; ++mt)
#pragma unroll
        for (int nt = 0; nt < 4; ++nt)
            acc[mt][nt] = (v8f){0.f,0.f,0.f,0.f,0.f,0.f,0.f,0.f};

    const unsigned short* aP[4];
    const unsigned short* bP[4];
#pragma unroll
    for (int mt = 0; mt < 4; ++mt) aP[mt] = predbf + (size_t)(mBase + mt * 16) * EMB_DIM;
#pragma unroll
    for (int nt = 0; nt < 4; ++nt) bP[nt] = tabbf + (size_t)(nBase + nt * 16) * EMB_DIM;

    for (int k = 0; k < EMB_DIM; k += 32) {
        v16bf a[4], b[4];
#pragma unroll
        for (int mt = 0; mt < 4; ++mt) a[mt] = load_frag_a(aP[mt], EMB_DIM, k, lane);
#pragma unroll
        for (int nt = 0; nt < 4; ++nt) b[nt] = load_frag_b(bP[nt], EMB_DIM, k, lane);
#pragma unroll
        for (int mt = 0; mt < 4; ++mt)
#pragma unroll
            for (int nt = 0; nt < 4; ++nt)
                acc[mt][nt] = __builtin_amdgcn_wmma_f32_16x16x32_bf16(
                    false, a[mt], false, b[nt], (short)0, acc[mt][nt], false, false);
    }

    // per-row argmax over this 64m x 64n wave tile, merged globally via atomic max
    const int half = lane >> 4, l15 = lane & 15;
#pragma unroll
    for (int mt = 0; mt < 4; ++mt) {
#pragma unroll
        for (int r = 0; r < 8; ++r) {
            float v = acc[mt][0][r];
            int   n = nBase + l15;
#pragma unroll
            for (int nt = 1; nt < 4; ++nt) {
                float cv = acc[mt][nt][r];
                int   cn = nBase + nt * 16 + l15;   // cn > n always, so strict > keeps lower idx on ties
                if (cv > v) { v = cv; n = cn; }
            }
            // reduce across the 16 lanes of this half (rows m=r vs m=r+8 are disjoint halves)
            for (int off = 8; off > 0; off >>= 1) {
                float ov = __shfl_xor(v, off, 16);
                int   on = __shfl_xor(n, off, 16);
                if (ov > v || (ov == v && on < n)) { v = ov; n = on; }
            }
            if (l15 == 0) {
                const int m = mBase + mt * 16 + half * 8 + r;
                unsigned u   = __builtin_bit_cast(unsigned int, v);
                unsigned key = (u & 0x80000000u) ? ~u : (u | 0x80000000u); // monotone f32->u32
                // low 32 bits = ~idx so that max() prefers the SMALLER index on ties
                unsigned long long pk = ((unsigned long long)key << 32) | (unsigned)(~(unsigned)n);
                atomicMax(amax + m, pk);
            }
        }
    }
}

// ---------------- gather + per-row loss partial ----------------
__global__ __launch_bounds__(256) void gather_loss(const unsigned long long* __restrict__ amax,
                                                   const float* __restrict__ table,
                                                   const float* __restrict__ predf,
                                                   float* __restrict__ out,
                                                   float* __restrict__ partial) {
    const int b   = blockIdx.x;
    const int tid = threadIdx.x;
    const unsigned low = (unsigned)(amax[b] & 0xffffffffull);
    const int idx = (int)(~low);
    const float* trow = table + (size_t)idx * EMB_DIM;
    const float* prow = predf + (size_t)b * EMB_DIM;
    float* orow = out + (size_t)b * EMB_DIM;
    float s = 0.f;
    for (int e = tid; e < EMB_DIM; e += 256) {
        float t = trow[e];
        float p = prow[e];
        orow[e] = t;
        float d = p - t;
        s += d * d;
    }
    __shared__ float red[256];
    red[tid] = s; __syncthreads();
    for (int st = 128; st > 0; st >>= 1) {
        if (tid < st) red[tid] += red[tid + st];
        __syncthreads();
    }
    if (tid == 0) partial[b] = 0.5f * red[0];
}

__global__ __launch_bounds__(256) void loss_reduce(const float* __restrict__ partial,
                                                   float* __restrict__ lossA,
                                                   float* __restrict__ lossB) {
    const int tid = threadIdx.x;
    float s = 0.f;
    for (int i = tid; i < B_DIM; i += 256) s += partial[i];
    __shared__ float red[256];
    red[tid] = s; __syncthreads();
    for (int st = 128; st > 0; st >>= 1) {
        if (tid < st) red[tid] += red[tid + st];
        __syncthreads();
    }
    if (tid == 0) { *lossA = red[0]; *lossB = red[0]; }
}

extern "C" void kernel_launch(void* const* d_in, const int* in_sizes, int n_in,
                              void* d_out, int out_size, void* d_ws, size_t ws_size,
                              hipStream_t stream) {
    const float* raw   = (const float*)d_in[0];   // [8192, 768]
    const float* W     = (const float*)d_in[1];   // [768, 3072]
    const float* bias  = (const float*)d_in[2];   // [3072]
    const float* table = (const float*)d_in[3];   // [8192, 3072]
    float* out = (float*)d_out;                   // [8192*3072] + lossA + lossB

    char* ws = (char*)d_ws;
    size_t off = 0;
    float*              predf   = (float*)(ws + off);              off += (size_t)B_DIM * EMB_DIM * 4;
    unsigned short*     predbf  = (unsigned short*)(ws + off);     off += (size_t)B_DIM * EMB_DIM * 2;
    unsigned short*     rawbf   = (unsigned short*)(ws + off);     off += (size_t)B_DIM * IN_DIM * 2;
    unsigned short*     wtbf    = (unsigned short*)(ws + off);     off += (size_t)EMB_DIM * IN_DIM * 2;
    unsigned short*     tabbf   = (unsigned short*)(ws + off);     off += (size_t)V_DIM * EMB_DIM * 2;
    unsigned long long* amax    = (unsigned long long*)(ws + off); off += (size_t)B_DIM * 8;
    float*              partial = (float*)(ws + off);              off += (size_t)B_DIM * 4;

    float* lossA = out + (size_t)B_DIM * EMB_DIM;
    float* lossB = lossA + 1;

    // down-convert to bf16 (table fits in 192MB L2 afterwards)
    cvt_bf16<<<(B_DIM * IN_DIM + 255) / 256, 256, 0, stream>>>(raw, rawbf, B_DIM * IN_DIM);
    cvt_bf16<<<(V_DIM * EMB_DIM + 255) / 256, 256, 0, stream>>>(table, tabbf, V_DIM * EMB_DIM);
    cvt_w_transpose<<<(IN_DIM * EMB_DIM + 255) / 256, 256, 0, stream>>>(W, wtbf);
    init_amax<<<(B_DIM + 255) / 256, 256, 0, stream>>>(amax);

    // GEMM1: 8192x3072x768
    gemm1_bias<<<dim3(B_DIM / 128, EMB_DIM / 128), 256, 0, stream>>>(rawbf, wtbf, bias, predf, predbf);
    // GEMM2 + argmax: 8192x8192x3072
    gemm2_argmax<<<dim3(B_DIM / 128, V_DIM / 256), 256, 0, stream>>>(predbf, tabbf, amax);
    // gather + deterministic loss reduction
    gather_loss<<<B_DIM, 256, 0, stream>>>(amax, table, predf, out, partial);
    loss_reduce<<<1, 256, 0, stream>>>(partial, lossA, lossB);
}
